// QLlamaDecoderLayer_25572235281039
// MI455X (gfx1250) — compile-verified, matrix operand order checked by reading, not run on hardware
//
#include <hip/hip_runtime.h>
#include <hip/hip_bf16.h>
#include <math.h>
#include <stdint.h>

// ---------------------------------------------------------------------------
// QLlamaDecoderLayer for MI455X (gfx1250, wave32, WMMA f16)
// GEMMs use async LDS-DMA double buffering (ASYNCcnt) + v_wmma_f32_16x16x32_f16.
// ---------------------------------------------------------------------------

typedef _Float16 v16h __attribute__((ext_vector_type(16)));
typedef float    v8f  __attribute__((ext_vector_type(8)));

#define S_LEN  2048
#define HID    2048
#define FF     5632
#define NH     16
#define NKV    8
#define HD     128
#define ATTN_SCALE 0.0883883476483184f   // 1/sqrt(128)

// ---------------- device helpers ----------------

__device__ __forceinline__ v16h make_frag(const _Float16* p0, const _Float16* p1) {
  union { v16h v; int4 i[2]; } u;
  u.i[0] = *reinterpret_cast<const int4*>(p0);
  u.i[1] = *reinterpret_cast<const int4*>(p1);
  return u.v;
}

__device__ __forceinline__ v8f wmma16(v16h a, v16h b, v8f c) {
  return __builtin_amdgcn_wmma_f32_16x16x32_f16(
      /*neg_a=*/false, a, /*neg_b=*/false, b,
      /*c_mod=*/(short)0, c, /*reuse_a=*/false, /*reuse_b=*/false);
}

// async 16B global -> LDS DMA (gfx1250), tracked by ASYNCcnt
__device__ __forceinline__ void async_copy_b128(unsigned lds_addr, const void* gaddr) {
  asm volatile("global_load_async_to_lds_b128 %0, %1, off"
               :: "v"(lds_addr), "v"(gaddr) : "memory");
}

__device__ __forceinline__ unsigned lds_addr_of(const void* p) {
  // generic pointers to LDS keep the byte offset in addr[31:0]
  return (unsigned)(uintptr_t)p;
}

// e2m1 magnitude rounding, matches searchsorted(bounds, a, 'left') semantics
__device__ __forceinline__ float e2m1_round(float a) {
  if (a > 5.0f)  return 6.0f;
  if (a > 3.5f)  return 4.0f;
  if (a > 2.5f)  return 3.0f;
  if (a > 1.75f) return 2.0f;
  if (a > 1.25f) return 1.5f;
  if (a > 0.75f) return 1.0f;
  if (a > 0.25f) return 0.5f;
  return 0.0f;
}

// ---------------- small utility kernels ----------------

__global__ void init_scalars_kernel(unsigned* s) {
  if (threadIdx.x < 64) s[threadIdx.x] = 0u;
}

__global__ void rmsnorm_kernel(const float* __restrict__ x,
                               const float* __restrict__ w,
                               float* __restrict__ y, int H) {
  const int row = blockIdx.x;
  const float* xr = x + (size_t)row * H;
  float* yr = y + (size_t)row * H;
  __shared__ float red[256];
  float s = 0.f;
  for (int i = threadIdx.x; i < H; i += 256) { float v = xr[i]; s += v * v; }
  red[threadIdx.x] = s;
  __syncthreads();
  for (int o = 128; o > 0; o >>= 1) {
    if (threadIdx.x < o) red[threadIdx.x] += red[threadIdx.x + o];
    __syncthreads();
  }
  const float inv = rsqrtf(red[0] / (float)H + 1e-5f);
  for (int i = threadIdx.x; i < H; i += 256) yr[i] = xr[i] * inv * w[i];
}

__global__ void absmax_kernel(const float* __restrict__ x, size_t n,
                              unsigned* __restrict__ slot) {
  unsigned m = 0u;
  const size_t stride = (size_t)gridDim.x * blockDim.x;
  for (size_t j = (size_t)blockIdx.x * blockDim.x + threadIdx.x; j < n; j += stride) {
    unsigned b = __float_as_uint(fabsf(x[j]));
    m = m > b ? m : b;
  }
  __shared__ unsigned red[256];
  red[threadIdx.x] = m;
  __syncthreads();
  for (int o = 128; o > 0; o >>= 1) {
    if (threadIdx.x < o) red[threadIdx.x] = max(red[threadIdx.x], red[threadIdx.x + o]);
    __syncthreads();
  }
  if (threadIdx.x == 0) atomicMax(slot, red[0]);
}

// NVFP4 fake-quant + channel reorder: dst[r][c] = fq(src[r][idx[c]]) as f16.
__global__ void quant_reorder_kernel(const float* __restrict__ src,
                                     const int* __restrict__ idx,
                                     const unsigned* __restrict__ gslot,
                                     _Float16* __restrict__ dst,
                                     int rows, int cols, int ldsrc) {
  const int gpr = cols >> 4;
  const int g = blockIdx.x * blockDim.x + threadIdx.x;
  if (g >= rows * gpr) return;
  const int row = g / gpr;
  const int grp = g - row * gpr;
  const float gmax = __uint_as_float(*gslot);
  const float gscale = fmaxf(gmax / 2688.0f, 1e-12f);  // 448*6
  const float* srow = src + (size_t)row * ldsrc;
  const int* ip = idx + grp * 16;
  float v[16];
  float amax = 0.f;
#pragma unroll
  for (int j = 0; j < 16; ++j) {
    v[j] = srow[ip[j]] / gscale;
    amax = fmaxf(amax, fabsf(v[j]));
  }
  const float sc = fmaxf(amax / 6.0f, 1e-12f);
  _Float16* drow = dst + (size_t)row * cols + grp * 16;
#pragma unroll
  for (int j = 0; j < 16; ++j) {
    float q = e2m1_round(fabsf(v[j]) / sc);
    drow[j] = (_Float16)(copysignf(q * sc, v[j]) * gscale);
  }
}

// ---------------- tiled WMMA GEMM: C[M,N] = A[M,K] * Bt[N,K]^T ----------------
// Block tile 64x256, 8 waves (2x4), wave tile 32x64, K-step 32.
// Double-buffered LDS tiles filled by async LDS-DMA (GLOBAL_LOAD_ASYNC_TO_LDS).

#define BM 64
#define BN 256
#define BK 32
#define LDSK (BK + 8)   // pad to 40 halves (80B rows, 16B aligned)

__launch_bounds__(256)
__global__ void gemm_f16_wmma(const _Float16* __restrict__ A,
                              const _Float16* __restrict__ Bt,
                              float* __restrict__ C,
                              int M, int N, int K) {
  __shared__ _Float16 As[2][BM][LDSK];
  __shared__ _Float16 Bs[2][BN][LDSK];

  const int tid = threadIdx.x;
  const int lane = tid & 31;
  const int wave = tid >> 5;
  const int waveM = wave >> 2;   // 0..1
  const int waveN = wave & 3;    // 0..3
  const int mBase = blockIdx.y * BM;
  const int nBase = blockIdx.x * BN;

  const int laneMod  = lane & 15;
  const int laneHalf = lane >> 4;
  const int kSel = laneHalf * 8;

  const v8f zf = {0.f, 0.f, 0.f, 0.f, 0.f, 0.f, 0.f, 0.f};
  v8f acc[2][4];
#pragma unroll
  for (int mi = 0; mi < 2; ++mi)
#pragma unroll
    for (int ni = 0; ni < 4; ++ni) acc[mi][ni] = zf;

  const int aRow = tid >> 2;            // 0..63
  const int aChunk = (tid & 3) * 8;     // 0,8,16,24
  const _Float16* gA = A + (size_t)(mBase + aRow) * K + aChunk;
  const _Float16* gB = Bt + (size_t)(nBase + tid) * K;

  // per-thread async ops per tile: 1 (A) + 4 (B) = 5
  auto issue_tile = [&](int t) {
    const int k0 = t * BK;
    const int b = t & 1;
    async_copy_b128(lds_addr_of(&As[b][aRow][aChunk]), gA + k0);
    const unsigned ldsB = lds_addr_of(&Bs[b][tid][0]);
    const _Float16* gb = gB + k0;
#pragma unroll
    for (int i = 0; i < 4; ++i)
      async_copy_b128(ldsB + 16u * i, gb + 8 * i);
  };

  const int ntiles = K / BK;
  issue_tile(0);

  for (int t = 0; t < ntiles; ++t) {
    if (t + 1 < ntiles) {
      issue_tile(t + 1);
      // drain only tile t's 5 ops (async loads complete in order)
      asm volatile("s_wait_asynccnt 0x5" ::: "memory");
    } else {
      asm volatile("s_wait_asynccnt 0x0" ::: "memory");
    }
    __syncthreads();   // tile t visible to all waves

    const int b = t & 1;
    v16h afr[2], bfr[4];
#pragma unroll
    for (int mi = 0; mi < 2; ++mi) {
      const _Float16* p = &As[b][waveM * 32 + mi * 16 + laneMod][kSel];
      afr[mi] = make_frag(p, p + 16);
    }
#pragma unroll
    for (int ni = 0; ni < 4; ++ni) {
      const _Float16* p = &Bs[b][waveN * 64 + ni * 16 + laneMod][kSel];
      bfr[ni] = make_frag(p, p + 16);
    }
#pragma unroll
    for (int mi = 0; mi < 2; ++mi)
#pragma unroll
      for (int ni = 0; ni < 4; ++ni)
        acc[mi][ni] = wmma16(afr[mi], bfr[ni], acc[mi][ni]);

    __syncthreads();   // all waves done with buffer b before it is re-filled
  }

#pragma unroll
  for (int mi = 0; mi < 2; ++mi)
#pragma unroll
    for (int ni = 0; ni < 4; ++ni) {
      const int row0 = mBase + waveM * 32 + mi * 16 + laneHalf * 8;
      const int col = nBase + waveN * 64 + ni * 16 + laneMod;
      float* cp = C + (size_t)row0 * N + col;
#pragma unroll
      for (int r = 0; r < 8; ++r) cp[(size_t)r * N] = acc[mi][ni][r];
    }
}

// ---------------- qkv post-processing ----------------

// K path: int4 asymmetric group(64) fake-quant, then RoPE; writes f16 [kv][s][d]
__launch_bounds__(128)
__global__ void kpost_kernel(const float* __restrict__ kf32,
                             const int* __restrict__ pos_ids,
                             _Float16* __restrict__ kf16, int S) {
  const int kv = blockIdx.x, s = blockIdx.y, d = threadIdx.x;
  const float val = kf32[(size_t)s * (NKV * HD) + kv * HD + d];
  __shared__ float redmax[128], redmin[128], sh[128];
  redmax[d] = val; redmin[d] = val;
  __syncthreads();
  for (int o = 32; o > 0; o >>= 1) {
    if ((d & 63) < o) {
      redmax[d] = fmaxf(redmax[d], redmax[d + o]);
      redmin[d] = fminf(redmin[d], redmin[d + o]);
    }
    __syncthreads();
  }
  const int gb = d & 64;
  const float wmax = redmax[gb], wmin = redmin[gb];
  const float scale = fmaxf(wmax - wmin, 1e-5f) / 15.0f;
  const float base = fminf(fmaxf(rintf(-wmin / scale), 0.f), 15.f);
  const float q = fminf(fmaxf(rintf(val / scale) + base, 0.f), 15.f);
  const float kq = (q - base) * scale;
  sh[d] = kq;
  __syncthreads();
  const float pos = (float)pos_ids[s];
  const float invf = __powf(10000.0f, -(float)((d & 63) * 2) / (float)HD);
  const float ang = pos * invf;
  const float rot = (d < 64) ? -sh[d + 64] : sh[d - 64];
  kf16[((size_t)kv * S + s) * HD + d] = (_Float16)(kq * __cosf(ang) + rot * __sinf(ang));
}

// Q path: RoPE only; writes f16 [h][s][d]
__launch_bounds__(128)
__global__ void qpost_kernel(const float* __restrict__ qf32,
                             const int* __restrict__ pos_ids,
                             _Float16* __restrict__ qf16, int S) {
  const int h = blockIdx.x, s = blockIdx.y, d = threadIdx.x;
  const float* row = qf32 + (size_t)s * HID + h * HD;
  const float val = row[d];
  const float other = (d < 64) ? -row[d + 64] : row[d - 64];
  const float pos = (float)pos_ids[s];
  const float invf = __powf(10000.0f, -(float)((d & 63) * 2) / (float)HD);
  const float ang = pos * invf;
  qf16[((size_t)h * S + s) * HD + d] = (_Float16)(val * __cosf(ang) + other * __sinf(ang));
}

// V transpose: f16 [kv][d][s]
__global__ void vtrans_kernel(const float* __restrict__ vf32,
                              _Float16* __restrict__ vt, int S) {
  const size_t total = (size_t)NKV * HD * S;
  const size_t t = (size_t)blockIdx.x * blockDim.x + threadIdx.x;
  if (t >= total) return;
  const int s = (int)(t % S);
  const size_t rest = t / S;
  const int d = (int)(rest % HD);
  const int kv = (int)(rest / HD);
  vt[t] = (_Float16)vf32[(size_t)s * (NKV * HD) + kv * HD + d];
}

// ---------------- flash attention (WMMA, online softmax) ----------------
// grid (NH, S/64), 128 threads = 4 waves; wave handles 16 query rows.

__launch_bounds__(128)
__global__ void attn_kernel(const _Float16* __restrict__ qf,
                            const _Float16* __restrict__ kf,
                            const _Float16* __restrict__ vt,
                            float* __restrict__ out, int S) {
  __shared__ _Float16 Pl[4][16][32];   // per-wave P staging
  const int head = blockIdx.x;
  const int kvh = head >> 1;           // n_rep = 2
  const int wave = threadIdx.x >> 5;
  const int lane = threadIdx.x & 31;
  const int laneMod = lane & 15, laneHalf = lane >> 4;
  const int q0 = blockIdx.y * 64 + wave * 16;

  const _Float16* qrow = qf + ((size_t)head * S + q0 + laneMod) * HD;
  v16h qfr[4];
#pragma unroll
  for (int c = 0; c < 4; ++c) {
    const int f0 = c * 32 + laneHalf * 8;
    qfr[c] = make_frag(qrow + f0, qrow + f0 + 16);
  }

  const v8f zf = {0.f, 0.f, 0.f, 0.f, 0.f, 0.f, 0.f, 0.f};
  v8f o[8];
  float m[8], l[8];
#pragma unroll
  for (int f = 0; f < 8; ++f) o[f] = zf;
#pragma unroll
  for (int r = 0; r < 8; ++r) { m[r] = -1e30f; l[r] = 0.f; }

  const int nkb = (q0 + 16 + 31) >> 5;
  for (int kb = 0; kb < nkb; ++kb) {
    const int kBase = kb * 32;

    v8f sacc[2] = {zf, zf};
#pragma unroll
    for (int c = 0; c < 4; ++c) {
      const int f0 = c * 32 + laneHalf * 8;
      const _Float16* kr0 = kf + ((size_t)kvh * S + kBase + laneMod) * HD + f0;
      const _Float16* kr1 = kr0 + 16 * HD;
      sacc[0] = wmma16(qfr[c], make_frag(kr0, kr0 + 16), sacc[0]);
      sacc[1] = wmma16(qfr[c], make_frag(kr1, kr1 + 16), sacc[1]);
    }

    float pv0[8], pv1[8];
#pragma unroll
    for (int r = 0; r < 8; ++r) {
      const int qg = q0 + r + laneHalf * 8;
      float s0 = sacc[0][r] * ATTN_SCALE;
      float s1 = sacc[1][r] * ATTN_SCALE;
      if (kBase + laneMod > qg)      s0 = -1e30f;
      if (kBase + 16 + laneMod > qg) s1 = -1e30f;
      float mt = fmaxf(s0, s1);
      mt = fmaxf(mt, __shfl_xor(mt, 1, 32));
      mt = fmaxf(mt, __shfl_xor(mt, 2, 32));
      mt = fmaxf(mt, __shfl_xor(mt, 4, 32));
      mt = fmaxf(mt, __shfl_xor(mt, 8, 32));
      const float mn = fmaxf(m[r], mt);
      const float corr = __expf(m[r] - mn);
      const float p0 = __expf(s0 - mn);
      const float p1 = __expf(s1 - mn);
      float ps = p0 + p1;
      ps += __shfl_xor(ps, 1, 32);
      ps += __shfl_xor(ps, 2, 32);
      ps += __shfl_xor(ps, 4, 32);
      ps += __shfl_xor(ps, 8, 32);
      l[r] = l[r] * corr + ps;
      m[r] = mn;
      pv0[r] = p0; pv1[r] = p1;
#pragma unroll
      for (int f = 0; f < 8; ++f) o[f][r] = o[f][r] * corr;
    }

    // stage P as f16 in A-fragment order (per-wave LDS slice; DS in-order per wave)
#pragma unroll
    for (int r = 0; r < 8; ++r) {
      Pl[wave][r + laneHalf * 8][laneMod]      = (_Float16)pv0[r];
      Pl[wave][r + laneHalf * 8][16 + laneMod] = (_Float16)pv1[r];
    }
    const _Float16* pp = &Pl[wave][laneMod][laneHalf * 8];
    const v16h pfr = make_frag(pp, pp + 16);

#pragma unroll
    for (int f = 0; f < 8; ++f) {
      const _Float16* vr =
          vt + ((size_t)kvh * HD + f * 16 + laneMod) * S + kBase + laneHalf * 8;
      o[f] = wmma16(pfr, make_frag(vr, vr + 16), o[f]);
    }
  }

#pragma unroll
  for (int f = 0; f < 8; ++f) {
#pragma unroll
    for (int r = 0; r < 8; ++r) {
      const int row = q0 + r + laneHalf * 8;
      const int col = head * HD + f * 16 + laneMod;
      out[(size_t)row * HID + col] = o[f][r] / l[r];
    }
  }
}

// ---------------- elementwise ----------------

__global__ void silu_mul_kernel(float* __restrict__ g,
                                const float* __restrict__ u, size_t n) {
  const size_t i = (size_t)blockIdx.x * blockDim.x + threadIdx.x;
  if (i >= n) return;
  const float x = g[i];
  g[i] = x / (1.f + __expf(-x)) * u[i];
}

__global__ void add_kernel(const float* __restrict__ a,
                           const float* __restrict__ b,
                           float* __restrict__ y, size_t n) {
  const size_t i = (size_t)blockIdx.x * blockDim.x + threadIdx.x;
  if (i >= n) return;
  y[i] = a[i] + b[i];
}

// ---------------- host orchestration ----------------

extern "C" void kernel_launch(void* const* d_in, const int* in_sizes, int n_in,
                              void* d_out, int out_size, void* d_ws, size_t ws_size,
                              hipStream_t stream) {
  (void)in_sizes; (void)n_in; (void)out_size; (void)ws_size;
  const float* hidden = (const float*)d_in[0];
  const int*   pos    = (const int*)d_in[1];
  const float* attn_w = (const float*)d_in[2];
  const float* mlp_w  = (const float*)d_in[3];
  const float* wq = (const float*)d_in[4];
  const float* wk = (const float*)d_in[5];
  const float* wv = (const float*)d_in[6];
  const float* wo = (const float*)d_in[7];
  const float* wg = (const float*)d_in[8];
  const float* wu = (const float*)d_in[9];
  const float* wd = (const float*)d_in[10];
  const int* idx_qkv  = (const int*)d_in[11];
  const int* idx_o    = (const int*)d_in[12];
  const int* idx_mlp  = (const int*)d_in[13];
  const int* idx_down = (const int*)d_in[14];

  const size_t SZ_SH_F  = (size_t)S_LEN * HID * 4;
  const size_t SZ_SHalf = (size_t)S_LEN * (HID / 2) * 4;
  const size_t SZ_SFF_H = (size_t)S_LEN * FF * 2;
  const size_t SZ_SFF_F = (size_t)S_LEN * FF * 4;

  char* w = (char*)d_ws;
  size_t off = 0;
  unsigned* scalars = (unsigned*)(w + off); off += 1024;
  float*    xn      = (float*)(w + off);    off += SZ_SH_F;
  _Float16* actq    = (_Float16*)(w + off); off += SZ_SFF_H;
  _Float16* w16     = (_Float16*)(w + off); off += SZ_SFF_H;
  float*    qbuf    = (float*)(w + off);    off += SZ_SH_F;    // also attn_out, d_mlp
  float*    kbuf    = (float*)(w + off);    off += SZ_SHalf;
  float*    vbuf    = (float*)(w + off);    off += SZ_SHalf;
  _Float16* q16     = (_Float16*)(w + off); off += (size_t)NH * S_LEN * HD * 2;
  _Float16* k16     = (_Float16*)(w + off); off += (size_t)NKV * S_LEN * HD * 2;
  _Float16* vt16    = (_Float16*)(w + off); off += (size_t)NKV * HD * S_LEN * 2;
  float*    hbuf    = (float*)(w + off);    off += SZ_SH_F;
  float*    gbuf    = (float*)(w + off);    off += SZ_SFF_F;   // also o-proj temp
  float*    ubuf    = (float*)(w + off);    off += SZ_SFF_F;

  const size_t nSH  = (size_t)S_LEN * HID;
  const size_t nSFF = (size_t)S_LEN * FF;

  auto qgrid = [](int rows, int cols) {
    return dim3((unsigned)((rows * (cols / 16) + 255) / 256));
  };
  const dim3 ggemm_h(HID / BN, S_LEN / BM);
  const dim3 ggemm_kv((HID / 2) / BN, S_LEN / BM);
  const dim3 ggemm_ff(FF / BN, S_LEN / BM);

  init_scalars_kernel<<<1, 64, 0, stream>>>(scalars);

  // ---- attention block ----
  rmsnorm_kernel<<<S_LEN, 256, 0, stream>>>(hidden, attn_w, xn, HID);
  absmax_kernel<<<512, 256, 0, stream>>>(xn, nSH, scalars + 0);
  quant_reorder_kernel<<<qgrid(S_LEN, HID), 256, 0, stream>>>(xn, idx_qkv, scalars + 0, actq, S_LEN, HID, HID);

  absmax_kernel<<<512, 256, 0, stream>>>(wq, (size_t)HID * HID, scalars + 1);
  quant_reorder_kernel<<<qgrid(HID, HID), 256, 0, stream>>>(wq, idx_qkv, scalars + 1, w16, HID, HID, HID);
  gemm_f16_wmma<<<ggemm_h, 256, 0, stream>>>(actq, w16, qbuf, S_LEN, HID, HID);

  absmax_kernel<<<512, 256, 0, stream>>>(wk, (size_t)(HID / 2) * HID, scalars + 2);
  quant_reorder_kernel<<<qgrid(HID / 2, HID), 256, 0, stream>>>(wk, idx_qkv, scalars + 2, w16, HID / 2, HID, HID);
  gemm_f16_wmma<<<ggemm_kv, 256, 0, stream>>>(actq, w16, kbuf, S_LEN, HID / 2, HID);

  absmax_kernel<<<512, 256, 0, stream>>>(wv, (size_t)(HID / 2) * HID, scalars + 3);
  quant_reorder_kernel<<<qgrid(HID / 2, HID), 256, 0, stream>>>(wv, idx_qkv, scalars + 3, w16, HID / 2, HID, HID);
  gemm_f16_wmma<<<ggemm_kv, 256, 0, stream>>>(actq, w16, vbuf, S_LEN, HID / 2, HID);

  kpost_kernel<<<dim3(NKV, S_LEN), 128, 0, stream>>>(kbuf, pos, k16, S_LEN);
  qpost_kernel<<<dim3(NH, S_LEN), 128, 0, stream>>>(qbuf, pos, q16, S_LEN);
  vtrans_kernel<<<(unsigned)(((size_t)NKV * HD * S_LEN + 255) / 256), 256, 0, stream>>>(vbuf, vt16, S_LEN);

  attn_kernel<<<dim3(NH, S_LEN / 64), 128, 0, stream>>>(q16, k16, vt16, qbuf, S_LEN);

  absmax_kernel<<<512, 256, 0, stream>>>(qbuf, nSH, scalars + 4);
  quant_reorder_kernel<<<qgrid(S_LEN, HID), 256, 0, stream>>>(qbuf, idx_o, scalars + 4, actq, S_LEN, HID, HID);
  absmax_kernel<<<512, 256, 0, stream>>>(wo, (size_t)HID * HID, scalars + 5);
  quant_reorder_kernel<<<qgrid(HID, HID), 256, 0, stream>>>(wo, idx_o, scalars + 5, w16, HID, HID, HID);
  gemm_f16_wmma<<<ggemm_h, 256, 0, stream>>>(actq, w16, gbuf, S_LEN, HID, HID);
  add_kernel<<<(unsigned)((nSH + 255) / 256), 256, 0, stream>>>(hidden, gbuf, hbuf, nSH);

  // ---- MLP block ----
  rmsnorm_kernel<<<S_LEN, 256, 0, stream>>>(hbuf, mlp_w, xn, HID);
  absmax_kernel<<<512, 256, 0, stream>>>(xn, nSH, scalars + 6);
  quant_reorder_kernel<<<qgrid(S_LEN, HID), 256, 0, stream>>>(xn, idx_mlp, scalars + 6, actq, S_LEN, HID, HID);

  absmax_kernel<<<512, 256, 0, stream>>>(wg, (size_t)FF * HID, scalars + 7);
  quant_reorder_kernel<<<qgrid(FF, HID), 256, 0, stream>>>(wg, idx_mlp, scalars + 7, w16, FF, HID, HID);
  gemm_f16_wmma<<<ggemm_ff, 256, 0, stream>>>(actq, w16, gbuf, S_LEN, FF, HID);

  absmax_kernel<<<512, 256, 0, stream>>>(wu, (size_t)FF * HID, scalars + 8);
  quant_reorder_kernel<<<qgrid(FF, HID), 256, 0, stream>>>(wu, idx_mlp, scalars + 8, w16, FF, HID, HID);
  gemm_f16_wmma<<<ggemm_ff, 256, 0, stream>>>(actq, w16, ubuf, S_LEN, FF, HID);

  silu_mul_kernel<<<(unsigned)((nSFF + 255) / 256), 256, 0, stream>>>(gbuf, ubuf, nSFF);
  absmax_kernel<<<512, 256, 0, stream>>>(gbuf, nSFF, scalars + 9);
  quant_reorder_kernel<<<qgrid(S_LEN, FF), 256, 0, stream>>>(gbuf, idx_down, scalars + 9, actq, S_LEN, FF, FF);

  absmax_kernel<<<512, 256, 0, stream>>>(wd, (size_t)HID * FF, scalars + 10);
  quant_reorder_kernel<<<qgrid(HID, FF), 256, 0, stream>>>(wd, idx_down, scalars + 10, w16, HID, FF, FF);
  gemm_f16_wmma<<<ggemm_h, 256, 0, stream>>>(actq, w16, qbuf, S_LEN, HID, FF);

  add_kernel<<<(unsigned)((nSH + 255) / 256), 256, 0, stream>>>(hbuf, qbuf, (float*)d_out, nSH);
}
